// Explainer_72069551227428
// MI455X (gfx1250) — compile-verified, matrix operand order checked by reading, not run on hardware
//
#include <hip/hip_runtime.h>
#include <hip/hip_bf16.h>

typedef __attribute__((ext_vector_type(16))) _Float16 v16h;
typedef __attribute__((ext_vector_type(8)))  _Float16 v8h;
typedef __attribute__((ext_vector_type(8)))  float    v8f;

#define N_NODES 100000
#define N_EDGES 1600000
#define DIM     128
#define NGRAPH  64
#define BN_EPS  1e-5f
#define MTILES  (N_NODES / 16)   // 6250 exactly

// ---------------------------------------------------------------------------
// Weight pre-pack: W [128 x 128] row-major (k, n) -> per-wave B fragments for
// v_wmma_f32_16x16x32_f16.  Layout: Wp[((nt*4 + kt)*32 + lane)*16 + e], where
// lane<16 holds column n=nt*16+lane, K chunks {0..7,16..23}; lane>=16 holds
// same column range with K chunks {8..15,24..31} (per CDNA5 ISA 7.12.2).
// ---------------------------------------------------------------------------
__global__ void pack_w_kernel(const float* __restrict__ W, _Float16* __restrict__ Wp) {
    int idx = blockIdx.x * blockDim.x + threadIdx.x;   // 0..16383
    if (idx >= 8 * 4 * 32 * 16) return;
    int e    =  idx        & 15;
    int lane = (idx >> 4)  & 31;
    int kt   = (idx >> 9)  & 3;
    int nt   =  idx >> 11;
    int n    = nt * 16 + (lane & 15);
    int grp  = lane >> 4;
    int k    = kt * 32 + grp * 8 + ((e < 8) ? e : (8 + e));
    Wp[idx]  = (_Float16)W[k * DIM + n];
}

// agg = h  (vectorized copy)
__global__ void copy4_kernel(const float* __restrict__ src, float* __restrict__ dst, int n4) {
    int i = blockIdx.x * blockDim.x + threadIdx.x;
    if (i < n4) ((float4*)dst)[i] = ((const float4*)src)[i];
}

// agg[dst] += h[src] over all edges; 32 lanes per edge, 4 floats per lane.
__global__ void scatter_kernel(const float* __restrict__ H, const int* __restrict__ ei,
                               float* __restrict__ AGG) {
    long long idx = (long long)blockIdx.x * blockDim.x + threadIdx.x;
    if (idx >= (long long)N_EDGES * 32) return;
    int e = (int)(idx >> 5);
    int f = ((int)idx & 31) << 2;
    int s = ei[e];
    int d = ei[N_EDGES + e];
    const float4 v = *(const float4*)&H[(size_t)s * DIM + f];
    float* a = &AGG[(size_t)d * DIM + f];
    atomicAdd(a + 0, v.x);
    atomicAdd(a + 1, v.y);
    atomicAdd(a + 2, v.z);
    atomicAdd(a + 3, v.w);
}

// Load A fragment (16 f16) for one lane from f32 row pointer p = row + kt*32 + base
__device__ inline v16h load_afrag_f32(const float* __restrict__ p) {
    const float4 f0 = *(const float4*)(p);
    const float4 f1 = *(const float4*)(p + 4);
    const float4 f2 = *(const float4*)(p + 16);
    const float4 f3 = *(const float4*)(p + 20);
    v16h a;
    a[0]  = (_Float16)f0.x; a[1]  = (_Float16)f0.y; a[2]  = (_Float16)f0.z; a[3]  = (_Float16)f0.w;
    a[4]  = (_Float16)f1.x; a[5]  = (_Float16)f1.y; a[6]  = (_Float16)f1.z; a[7]  = (_Float16)f1.w;
    a[8]  = (_Float16)f2.x; a[9]  = (_Float16)f2.y; a[10] = (_Float16)f2.z; a[11] = (_Float16)f2.w;
    a[12] = (_Float16)f3.x; a[13] = (_Float16)f3.y; a[14] = (_Float16)f3.z; a[15] = (_Float16)f3.w;
    return a;
}

// ---------------------------------------------------------------------------
// Fused GIN MLP (hidden->hidden layers): Y = relu(Z@W1 + b1) @ W2 + b2
// One wave (32 threads) per 16-row tile.  GEMM1 -> relu -> LDS transpose
// (D-fragment layout -> A-fragment layout) -> GEMM2.
// ---------------------------------------------------------------------------
__global__ void __launch_bounds__(32)
gin_mlp128_kernel(const float* __restrict__ Z,
                  const _Float16* __restrict__ Wp1, const float* __restrict__ b1,
                  const _Float16* __restrict__ Wp2, const float* __restrict__ b2,
                  float* __restrict__ Y) {
    __shared__ __align__(16) _Float16 lds[16 * DIM];
    const int lane = threadIdx.x;
    const int ml   = lane & 15;
    const int grp  = lane >> 4;
    const int base = grp * 8;
    const int row0 = blockIdx.x * 16;

    v8f acc[8];
    for (int t = 0; t < 8; ++t)
        for (int r = 0; r < 8; ++r) acc[t][r] = 0.0f;

    // ---- GEMM1: Z @ W1 ----
    const float* zrow = Z + (size_t)(row0 + ml) * DIM;
    const v16h* wp1 = (const v16h*)Wp1;
    for (int kt = 0; kt < 4; ++kt) {
        v16h a = load_afrag_f32(zrow + kt * 32 + base);
        for (int nt = 0; nt < 8; ++nt) {
            v16h b = wp1[(nt * 4 + kt) * 32 + lane];
            acc[nt] = __builtin_amdgcn_wmma_f32_16x16x32_f16(
                false, a, false, b, (short)0, acc[nt], false, false);
        }
    }

    // bias + relu, stash f16 tile to LDS in [row][k] order
    for (int nt = 0; nt < 8; ++nt) {
        int c = nt * 16 + ml;
        float bb = b1[c];
        for (int r = 0; r < 8; ++r) {
            float v = acc[nt][r] + bb;
            v = v > 0.0f ? v : 0.0f;
            lds[(r + base) * DIM + c] = (_Float16)v;
        }
    }
    __syncthreads();

    // ---- GEMM2: h1 @ W2 ----
    for (int t = 0; t < 8; ++t)
        for (int r = 0; r < 8; ++r) acc[t][r] = 0.0f;
    const v16h* wp2 = (const v16h*)Wp2;
    for (int kt = 0; kt < 4; ++kt) {
        const _Float16* lp = &lds[ml * DIM + kt * 32 + base];
        v8h lo = *(const v8h*)lp;
        v8h hi = *(const v8h*)(lp + 16);
        v16h a;
        for (int i = 0; i < 8; ++i) { a[i] = lo[i]; a[8 + i] = hi[i]; }
        for (int nt = 0; nt < 8; ++nt) {
            v16h b = wp2[(nt * 4 + kt) * 32 + lane];
            acc[nt] = __builtin_amdgcn_wmma_f32_16x16x32_f16(
                false, a, false, b, (short)0, acc[nt], false, false);
        }
    }

    for (int nt = 0; nt < 8; ++nt) {
        int c = nt * 16 + ml;
        float bb = b2[c];
        for (int r = 0; r < 8; ++r)
            Y[(size_t)(row0 + base + r) * DIM + c] = acc[nt][r] + bb;
    }
}

// ---------------------------------------------------------------------------
// Final layer: Yf = relu(Z@W1 + b1) @ w2col + b2   (DOUT = 1)
// GEMM1 via WMMA; the length-128 dot collapses across lanes via shfl_xor.
// ---------------------------------------------------------------------------
__global__ void __launch_bounds__(32)
gin_mlp_final_kernel(const float* __restrict__ Z,
                     const _Float16* __restrict__ Wp1, const float* __restrict__ b1,
                     const float* __restrict__ w2, const float* __restrict__ b2,
                     float* __restrict__ Yf) {
    const int lane = threadIdx.x;
    const int ml   = lane & 15;
    const int grp  = lane >> 4;
    const int base = grp * 8;
    const int row0 = blockIdx.x * 16;

    v8f acc[8];
    for (int t = 0; t < 8; ++t)
        for (int r = 0; r < 8; ++r) acc[t][r] = 0.0f;

    const float* zrow = Z + (size_t)(row0 + ml) * DIM;
    const v16h* wp1 = (const v16h*)Wp1;
    for (int kt = 0; kt < 4; ++kt) {
        v16h a = load_afrag_f32(zrow + kt * 32 + base);
        for (int nt = 0; nt < 8; ++nt) {
            v16h b = wp1[(nt * 4 + kt) * 32 + lane];
            acc[nt] = __builtin_amdgcn_wmma_f32_16x16x32_f16(
                false, a, false, b, (short)0, acc[nt], false, false);
        }
    }

    float part[8] = {0, 0, 0, 0, 0, 0, 0, 0};
    for (int nt = 0; nt < 8; ++nt) {
        int c = nt * 16 + ml;
        float bb = b1[c], w = w2[c];
        for (int r = 0; r < 8; ++r) {
            float v = acc[nt][r] + bb;
            v = v > 0.0f ? v : 0.0f;
            part[r] += v * w;
        }
    }
    // reduce over the 16 lanes of each half-wave (columns)
    for (int m = 1; m < 16; m <<= 1)
        for (int r = 0; r < 8; ++r) part[r] += __shfl_xor(part[r], m, 32);
    if (ml == 0) {
        float bb = b2[0];
        for (int r = 0; r < 8; ++r) Yf[row0 + base + r] = part[r] + bb;
    }
}

// ---------------------------------------------------------------------------
// BatchNorm (training-mode, biased var) over N rows x 128 cols
// ---------------------------------------------------------------------------
__global__ void zero_kernel(float* __restrict__ p, int n) {
    int i = blockIdx.x * blockDim.x + threadIdx.x;
    if (i < n) p[i] = 0.0f;
}

__global__ void bn_stats_kernel(const float* __restrict__ Y, float* __restrict__ stats) {
    int c = threadIdx.x;                                    // 128 threads = 128 cols
    int rows = (N_NODES + gridDim.x - 1) / gridDim.x;
    int r0 = blockIdx.x * rows;
    int r1 = min(r0 + rows, N_NODES);
    float s = 0.0f, s2 = 0.0f;
    for (int r = r0; r < r1; ++r) {
        float v = Y[(size_t)r * DIM + c];
        s += v; s2 += v * v;
    }
    atomicAdd(&stats[c], s);
    atomicAdd(&stats[DIM + c], s2);
}

__global__ void bn_apply_kernel(const float* __restrict__ Y, const float* __restrict__ stats,
                                const float* __restrict__ gamma, const float* __restrict__ beta,
                                float* __restrict__ H, int do_relu) {
    long long idx = (long long)blockIdx.x * blockDim.x + threadIdx.x;
    if (idx >= (long long)N_NODES * DIM) return;
    int c = (int)idx & (DIM - 1);
    const float invn = 1.0f / (float)N_NODES;
    float mu  = stats[c] * invn;
    float var = stats[DIM + c] * invn - mu * mu;
    float v = gamma[c] * (Y[idx] - mu) * rsqrtf(var + BN_EPS) + beta[c];
    if (do_relu) v = fmaxf(v, 0.0f);
    H[idx] = v;
}

// single-column stats for the final [N,1] output
__global__ void fstats_kernel(const float* __restrict__ Yf, float* __restrict__ st2) {
    float s = 0.0f, s2 = 0.0f;
    for (int i = blockIdx.x * blockDim.x + threadIdx.x; i < N_NODES;
         i += gridDim.x * blockDim.x) {
        float v = Yf[i];
        s += v; s2 += v * v;
    }
    for (int m = 1; m < 32; m <<= 1) { s += __shfl_xor(s, m, 32); s2 += __shfl_xor(s2, m, 32); }
    if ((threadIdx.x & 31) == 0) { atomicAdd(&st2[0], s); atomicAdd(&st2[1], s2); }
}

// ---------------------------------------------------------------------------
// Segment softmax over sorted `batch` (G=64 graphs)
// ---------------------------------------------------------------------------
__device__ inline unsigned fmap(float f) {
    unsigned u = __float_as_uint(f);
    return (u & 0x80000000u) ? ~u : (u | 0x80000000u);   // order-preserving map
}
__device__ inline float funmap(unsigned u) {
    return __uint_as_float((u & 0x80000000u) ? (u & 0x7fffffffu) : ~u);
}

__global__ void final_bn_kernel(const float* __restrict__ Yf, const float* __restrict__ st2,
                                const float* __restrict__ gamma, const float* __restrict__ beta,
                                const int* __restrict__ batch,
                                float* __restrict__ T, unsigned* __restrict__ mbits) {
    int i = blockIdx.x * blockDim.x + threadIdx.x;
    if (i >= N_NODES) return;
    const float invn = 1.0f / (float)N_NODES;
    float mu  = st2[0] * invn;
    float var = st2[1] * invn - mu * mu;
    float z = gamma[0] * (Yf[i] - mu) * rsqrtf(var + BN_EPS) + beta[0];
    float t = z * 0.2f;                                   // h / 5
    T[i] = t;
    atomicMax(&mbits[batch[i]], fmap(t));
}

__global__ void softmax_exp_kernel(const float* __restrict__ T, const int* __restrict__ batch,
                                   const unsigned* __restrict__ mbits,
                                   float* __restrict__ ssum, float* __restrict__ out) {
    int i = blockIdx.x * blockDim.x + threadIdx.x;
    if (i >= N_NODES) return;
    int g = batch[i];
    float e = expf(T[i] - funmap(mbits[g]));
    out[i] = e;
    atomicAdd(&ssum[g], e);
}

__global__ void softmax_norm_kernel(float* __restrict__ out, const int* __restrict__ batch,
                                    const float* __restrict__ ssum) {
    int i = blockIdx.x * blockDim.x + threadIdx.x;
    if (i >= N_NODES) return;
    out[i] = out[i] / ssum[batch[i]];
}

// ---------------------------------------------------------------------------
extern "C" void kernel_launch(void* const* d_in, const int* in_sizes, int n_in,
                              void* d_out, int out_size, void* d_ws, size_t ws_size,
                              hipStream_t stream) {
    (void)in_sizes; (void)n_in; (void)out_size; (void)ws_size;

    const float* x     = (const float*)d_in[0];
    const int*   ei    = (const int*)d_in[1];      // [2, E]
    const int*   batch = (const int*)d_in[2];      // [N]
    const float *W1[3], *B1[3], *W2[3], *B2[3], *GA[3], *BE[3];
    for (int l = 0; l < 3; ++l) {
        W1[l] = (const float*)d_in[3 + l * 6 + 0];
        B1[l] = (const float*)d_in[3 + l * 6 + 1];
        W2[l] = (const float*)d_in[3 + l * 6 + 2];
        B2[l] = (const float*)d_in[3 + l * 6 + 3];
        GA[l] = (const float*)d_in[3 + l * 6 + 4];
        BE[l] = (const float*)d_in[3 + l * 6 + 5];
    }

    // workspace layout
    float*     ws    = (float*)d_ws;
    float*     agg   = ws;                                   // N*128
    float*     ybuf  = agg  + (size_t)N_NODES * DIM;         // N*128
    float*     hbuf  = ybuf + (size_t)N_NODES * DIM;         // N*128
    _Float16*  wp1   = (_Float16*)(hbuf + (size_t)N_NODES * DIM);  // 128*128 f16
    _Float16*  wp2   = wp1 + DIM * DIM;                      // 128*128 f16
    float*     stats = (float*)(wp2 + DIM * DIM);            // 256
    unsigned*  mbits = (unsigned*)(stats + 256);             // 64
    float*     ssum  = (float*)(mbits + NGRAPH);             // 64
    float*     yfin  = ybuf;   // final-layer scalar output reuses ybuf
    float*     tbuf  = agg;    // t = bn(y)/5 reuses agg (consumed by then)

    const int n4         = N_NODES * DIM / 4;
    const int copyGrid   = (n4 + 255) / 256;
    const long long sct  = (long long)N_EDGES * 32;
    const int scatterGrid = (int)((sct + 255) / 256);
    const int elemGrid   = (int)(((long long)N_NODES * DIM + 255) / 256);
    const int nodeGrid   = (N_NODES + 255) / 256;

    const float* h = x;
    for (int l = 0; l < 2; ++l) {
        copy4_kernel   <<<copyGrid, 256, 0, stream>>>(h, agg, n4);
        scatter_kernel <<<scatterGrid, 256, 0, stream>>>(h, ei, agg);
        pack_w_kernel  <<<64, 256, 0, stream>>>(W1[l], wp1);
        pack_w_kernel  <<<64, 256, 0, stream>>>(W2[l], wp2);
        gin_mlp128_kernel<<<MTILES, 32, 0, stream>>>(agg, wp1, B1[l], wp2, B2[l], ybuf);
        zero_kernel    <<<2, 256, 0, stream>>>(stats, 384);   // stats + mbits + ssum
        bn_stats_kernel<<<256, 128, 0, stream>>>(ybuf, stats);
        bn_apply_kernel<<<elemGrid, 256, 0, stream>>>(ybuf, stats, GA[l], BE[l], hbuf, 1);
        h = hbuf;
    }

    // layer 2 (projects to 1) + segment softmax
    copy4_kernel   <<<copyGrid, 256, 0, stream>>>(h, agg, n4);
    scatter_kernel <<<scatterGrid, 256, 0, stream>>>(h, ei, agg);
    pack_w_kernel  <<<64, 256, 0, stream>>>(W1[2], wp1);
    gin_mlp_final_kernel<<<MTILES, 32, 0, stream>>>(agg, wp1, B1[2], W2[2], B2[2], yfin);
    zero_kernel    <<<2, 256, 0, stream>>>(stats, 384);
    fstats_kernel  <<<128, 256, 0, stream>>>(yfin, stats);
    final_bn_kernel<<<nodeGrid, 256, 0, stream>>>(yfin, stats, GA[2], BE[2], batch, tbuf, mbits);
    softmax_exp_kernel <<<nodeGrid, 256, 0, stream>>>(tbuf, batch, mbits, ssum, (float*)d_out);
    softmax_norm_kernel<<<nodeGrid, 256, 0, stream>>>((float*)d_out, batch, ssum);
}
